// SelfLayer_47854525612135
// MI455X (gfx1250) — matrix operand order repaired and checked by reading, not compiled, not run on hardware
//
#include <hip/hip_runtime.h>

typedef __attribute__((ext_vector_type(16))) _Float16 v16h;
typedef __attribute__((ext_vector_type(8)))  float    v8f;
typedef __attribute__((ext_vector_type(8)))  unsigned v8u;
typedef __attribute__((ext_vector_type(4)))  unsigned u32x4;
typedef __attribute__((ext_vector_type(8)))  int      i32x8;
typedef __attribute__((ext_vector_type(4)))  int      i32x4;

union Frag { v16h h; v8u u; uint4 q[2]; };

#define BATCH 16
#define SEQ   2048
#define DH    64

#if __has_builtin(__builtin_amdgcn_tensor_load_to_lds)
#define HAVE_TDM 1
#else
#define HAVE_TDM 0
#endif

static __device__ __forceinline__ v8f wmma_f16(const Frag& a, const Frag& b, v8f c) {
    return __builtin_amdgcn_wmma_f32_16x16x32_f16(
        false, a.h, false, b.h, (short)0, c, false, false);
}

#if HAVE_TDM
// Build a 2D tensor-DMA descriptor (D#) and issue TENSOR_LOAD_TO_LDS.
// dims/strides in elements (data_size = 2 bytes, f16).
// 6-arg builtin (clang-23 / therock form): (g0, g1, g2, g3, g_extra, cpol).
static __device__ __forceinline__ void tdm_load_2d(
    unsigned lds_addr, const _Float16* gaddr,
    unsigned tensor_d0, unsigned tensor_d1,
    unsigned tile_d0, unsigned tile_d1, unsigned stride0)
{
    unsigned long long ga = (unsigned long long)(size_t)gaddr;
    u32x4 g0;
    g0[0] = 1u;                                        // count=1, user descriptor
    g0[1] = lds_addr;                                  // LDS byte address
    g0[2] = (unsigned)(ga & 0xFFFFFFFFu);              // global_addr[31:0]
    g0[3] = (unsigned)((ga >> 32) & 0x01FFFFFFu)       // global_addr[56:32]
          | (2u << 30);                                // type=2 (image)
    i32x8 g1;
    g1[0] = (int)(1u << 16);                           // data_size=1 (2B); mask/flags=0
    g1[1] = (int)((tensor_d0 & 0xFFFFu) << 16);        // tensor_dim0[15:0]
    g1[2] = (int)(((tensor_d0 >> 16) & 0xFFFFu)
          | ((tensor_d1 & 0xFFFFu) << 16));            // dim0[31:16] | dim1[15:0]
    g1[3] = (int)(((tensor_d1 >> 16) & 0xFFFFu)
          | ((tile_d0 & 0xFFFFu) << 16));              // dim1[31:16] | tile_dim0
    g1[4] = (int)(tile_d1 & 0xFFFFu);                  // tile_dim1 (tile_dim2=0)
    g1[5] = (int)stride0;                              // tensor_dim0_stride[31:0]
    g1[6] = 0;                                         // stride0 hi | stride1 lo
    g1[7] = 0;
    i32x4 gz4  = {0, 0, 0, 0};
    i32x8 gz8  = {0, 0, 0, 0, 0, 0, 0, 0};
    __builtin_amdgcn_tensor_load_to_lds(g0, g1, gz4, gz4, gz8, 0);
}
#endif

// ---------------------------------------------------------------------------
// Kernel 1: projections  X[b,s,64] @ W[64,64] -> f16
//   p=0: Q row-major   p=1: K row-major   p=2: V transposed Vt[b][d][s]
// W^T staged once per block into LDS as f16 -> fragment loads are b128.
// ---------------------------------------------------------------------------
__global__ __launch_bounds__(128) void proj_kernel(
    const float* __restrict__ q_in, const float* __restrict__ k_in,
    const float* __restrict__ v_in,
    const float* __restrict__ WQ, const float* __restrict__ WK,
    const float* __restrict__ WV,
    _Float16* __restrict__ Qh, _Float16* __restrict__ Kh,
    _Float16* __restrict__ Vt)
{
    __shared__ _Float16 Wt[64 * 64];        // [col][kk], 8KB

    const int wave = threadIdx.x >> 5;
    const int lane = threadIdx.x & 31;
    const int ln   = lane & 15;
    const int hi   = lane >> 4;

    int bid  = blockIdx.x;
    int p    = bid % 3;
    int rest = bid / 3;
    int b    = rest & (BATCH - 1);
    int t    = rest / BATCH;
    int row0 = t * 64 + wave * 16;

    const float* X = (p == 0) ? q_in : (p == 1) ? k_in : v_in;
    const float* W = (p == 0) ? WQ   : (p == 1) ? WK   : WV;

    // stage W^T (f16) cooperatively: thread -> one row kk, 32 cols
    {
        int kk  = threadIdx.x & 63;
        int sel = threadIdx.x >> 6;                    // 0 / 1
        const float* wrow = W + kk * DH + sel * 32;
        #pragma unroll
        for (int j = 0; j < 32; j += 4) {
            float4 w4 = *(const float4*)(wrow + j);
            int c = sel * 32 + j;
            Wt[(c + 0) * DH + kk] = (_Float16)w4.x;
            Wt[(c + 1) * DH + kk] = (_Float16)w4.y;
            Wt[(c + 2) * DH + kk] = (_Float16)w4.z;
            Wt[(c + 3) * DH + kk] = (_Float16)w4.w;
        }
    }
    __syncthreads();

    // A fragments: 16 rows of X (float4 loads, cvt to f16)
    Frag a[2];
    {
        const float* xrow = X + ((size_t)b * SEQ + row0 + ln) * DH;
        #pragma unroll
        for (int c = 0; c < 2; ++c)
            #pragma unroll
            for (int vv = 0; vv < 2; ++vv) {
                int kkb = c * 32 + hi * 8 + vv * 16;
                float4 f0 = *(const float4*)(xrow + kkb);
                float4 f1 = *(const float4*)(xrow + kkb + 4);
                a[c].h[vv * 8 + 0] = (_Float16)f0.x;
                a[c].h[vv * 8 + 1] = (_Float16)f0.y;
                a[c].h[vv * 8 + 2] = (_Float16)f0.z;
                a[c].h[vv * 8 + 3] = (_Float16)f0.w;
                a[c].h[vv * 8 + 4] = (_Float16)f1.x;
                a[c].h[vv * 8 + 5] = (_Float16)f1.y;
                a[c].h[vv * 8 + 6] = (_Float16)f1.z;
                a[c].h[vv * 8 + 7] = (_Float16)f1.w;
            }
    }

    // B fragments of W from LDS (b128 pairs)
    Frag wb[2][4];
    #pragma unroll
    for (int c = 0; c < 2; ++c)
        #pragma unroll
        for (int nb = 0; nb < 4; ++nb) {
            const _Float16* src = Wt + (nb * 16 + ln) * DH + c * 32 + 8 * hi;
            wb[c][nb].q[0] = *(const uint4*)(src);
            wb[c][nb].q[1] = *(const uint4*)(src + 16);
        }

    v8f acc[4] = {};
    #pragma unroll
    for (int nb = 0; nb < 4; ++nb)
        #pragma unroll
        for (int c = 0; c < 2; ++c)
            acc[nb] = wmma_f16(a[c], wb[c][nb], acc[nb]);

    if (p < 2) {
        _Float16* O = (p == 0) ? Qh : Kh;
        #pragma unroll
        for (int nb = 0; nb < 4; ++nb)
            #pragma unroll
            for (int r = 0; r < 8; ++r) {
                int m = r + 8 * hi;
                O[((size_t)b * SEQ + row0 + m) * DH + nb * 16 + ln] =
                    (_Float16)acc[nb][r];
            }
    } else {
        #pragma unroll
        for (int nb = 0; nb < 4; ++nb)
            #pragma unroll
            for (int r = 0; r < 8; ++r) {
                int m = r + 8 * hi;
                int d = nb * 16 + ln;
                Vt[((size_t)b * DH + d) * SEQ + row0 + m] = (_Float16)acc[nb][r];
            }
    }
}

// ---------------------------------------------------------------------------
// Kernel 2: flash-style attention (unstabilized exp == reference).
// Block = 4 waves x 16 queries, same batch -> K/V chunks staged ONCE per
// block in LDS, double-buffered via TDM (tensor_load_to_lds + tensorcnt).
// Rowsum via an extra WMMA against a ones-column B fragment.
// ---------------------------------------------------------------------------
__global__ __launch_bounds__(128) void attn_kernel(
    const _Float16* __restrict__ Qh, const _Float16* __restrict__ Kh,
    const _Float16* __restrict__ Vt, const float* __restrict__ mask,
    float* __restrict__ out)
{
    __shared__ _Float16 ldsK[2][32 * DH];   // [buf][key][kk]   8KB
    __shared__ _Float16 ldsV[2][DH * 32];   // [buf][d][key]    8KB
    __shared__ _Float16 pLds[4][16 * 32];   // per-wave P slice 4KB

    const int wave = threadIdx.x >> 5;
    const int lane = threadIdx.x & 31;
    const int ln   = lane & 15;
    const int hi   = lane >> 4;

    int bid = blockIdx.x;
    int b   = bid >> 5;
    int t   = bid & 31;
    int q0  = t * 64 + wave * 16;

    const _Float16* kbase = Kh + (size_t)b * SEQ * DH;
    const _Float16* vbase = Vt + (size_t)b * DH * SEQ;

    // Q A-fragments: b128 pair loads
    Frag qa[2];
    {
        const _Float16* qrow = Qh + ((size_t)b * SEQ + q0 + ln) * DH;
        #pragma unroll
        for (int c = 0; c < 2; ++c) {
            qa[c].q[0] = *(const uint4*)(qrow + c * 32 + hi * 8);
            qa[c].q[1] = *(const uint4*)(qrow + c * 32 + hi * 8 + 16);
        }
    }

    // ones-column B fragment -> rowsum WMMA
    Frag onesb;
    #pragma unroll
    for (int h = 0; h < 16; ++h)
        onesb.h[h] = (ln == 0) ? (_Float16)1.0f : (_Float16)0.0f;

    v8f oacc[4] = {};
    v8f lacc    = {};
    _Float16* pbuf = pLds[wave];
    const float rscale = 0.125f;

#if HAVE_TDM
    if (wave == 0) {
        tdm_load_2d((unsigned)(size_t)&ldsK[0][0], kbase, DH, SEQ, DH, 32, DH);
        tdm_load_2d((unsigned)(size_t)&ldsV[0][0], vbase, SEQ, DH, 32, DH, SEQ);
    }
#endif
    int buf = 0;

    for (int k0 = 0; k0 < SEQ; k0 += 32) {
#if HAVE_TDM
        __syncthreads();               // everyone done reading buf^1
        if (wave == 0) {
            if (k0 + 32 < SEQ) {
                tdm_load_2d((unsigned)(size_t)&ldsK[buf ^ 1][0],
                            kbase + (size_t)(k0 + 32) * DH, DH, SEQ, DH, 32, DH);
                tdm_load_2d((unsigned)(size_t)&ldsV[buf ^ 1][0],
                            vbase + (k0 + 32), SEQ, DH, 32, DH, SEQ);
                __builtin_amdgcn_s_wait_tensorcnt(2);   // current pair done
            } else {
                __builtin_amdgcn_s_wait_tensorcnt(0);
            }
        }
        __syncthreads();               // buf is valid for all waves
        const _Float16* curK = ldsK[buf];
        const _Float16* curV = ldsV[buf];
        buf ^= 1;
#else
        __syncthreads();
        {   // cooperative b128 copy of this chunk (fallback path)
            int tid = threadIdx.x;
            const uint4* gk = (const uint4*)(kbase + (size_t)k0 * DH);
            uint4* sk = (uint4*)&ldsK[0][0];
            sk[tid]       = gk[tid];
            sk[tid + 128] = gk[tid + 128];
            #pragma unroll
            for (int rep = 0; rep < 2; ++rep) {
                int i   = tid + rep * 128;      // 0..255
                int row = i >> 2, off = (i & 3) * 8;
                *(uint4*)&ldsV[0][row * 32 + off] =
                    *(const uint4*)(vbase + (size_t)row * SEQ + k0 + off);
            }
        }
        __syncthreads();
        const _Float16* curK = ldsK[0];
        const _Float16* curV = ldsV[0];
#endif
        // prefetch next mask rows
        if (k0 + 32 < SEQ)
            __builtin_prefetch(mask + ((size_t)b * SEQ + q0 + (lane & 15)) * SEQ
                                    + k0 + 32, 0, 1);

        // --- scores: Q @ K^T over this 32-key chunk ---
        v8f sc[2] = {};
        #pragma unroll
        for (int g = 0; g < 2; ++g)
            #pragma unroll
            for (int c = 0; c < 2; ++c) {
                Frag kb;
                const _Float16* kr = curK + (g * 16 + ln) * DH + c * 32 + 8 * hi;
                kb.q[0] = *(const uint4*)kr;
                kb.q[1] = *(const uint4*)(kr + 16);
                sc[g] = wmma_f16(qa[c], kb, sc[g]);
            }

        // --- P = exp(S/8) * mask -> f16 in private LDS slice ---
        #pragma unroll
        for (int g = 0; g < 2; ++g) {
            const float* mcol =
                mask + ((size_t)b * SEQ + q0) * SEQ + (k0 + g * 16 + ln);
            #pragma unroll
            for (int r = 0; r < 8; ++r) {
                int m = r + 8 * hi;
                float pv = __expf(sc[g][r] * rscale) * mcol[(size_t)m * SEQ];
                pbuf[m * 32 + g * 16 + ln] = (_Float16)pv;
            }
        }

        // --- reload P as one 16x32 A fragment (b128 pair, same-wave LDS) ---
        Frag pa;
        pa.q[0] = *(const uint4*)(pbuf + ln * 32 + hi * 8);
        pa.q[1] = *(const uint4*)(pbuf + ln * 32 + hi * 8 + 16);

        // --- P @ V (4 d-blocks) + rowsum WMMA ---
        #pragma unroll
        for (int nb = 0; nb < 4; ++nb) {
            Frag vb;
            const _Float16* vr = curV + (nb * 16 + ln) * 32 + 8 * hi;
            vb.q[0] = *(const uint4*)vr;
            vb.q[1] = *(const uint4*)(vr + 16);
            oacc[nb] = wmma_f16(pa, vb, oacc[nb]);
        }
        lacc = wmma_f16(pa, onesb, lacc);
    }

    // --- normalize and store ---
    #pragma unroll
    for (int r = 0; r < 8; ++r) {
        float l   = __shfl(lacc[r], hi * 16, 32);
        float inv = 1.0f / l;
        int   m   = r + 8 * hi;
        float* orow = out + ((size_t)b * SEQ + q0 + m) * DH + ln;
        #pragma unroll
        for (int nb = 0; nb < 4; ++nb)
            orow[nb * 16] = oacc[nb][r] * inv;
    }
}

// ---------------------------------------------------------------------------
extern "C" void kernel_launch(void* const* d_in, const int* in_sizes, int n_in,
                              void* d_out, int out_size, void* d_ws, size_t ws_size,
                              hipStream_t stream) {
    (void)in_sizes; (void)n_in; (void)out_size; (void)ws_size;
    const float* q_in = (const float*)d_in[0];
    const float* k_in = (const float*)d_in[1];
    const float* v_in = (const float*)d_in[2];
    const float* mask = (const float*)d_in[3];
    const float* WQ   = (const float*)d_in[4];
    const float* WK   = (const float*)d_in[5];
    const float* WV   = (const float*)d_in[6];
    float* out = (float*)d_out;

    const size_t elems = (size_t)BATCH * SEQ * DH;
    _Float16* Qh = (_Float16*)d_ws;
    _Float16* Kh = Qh + elems;
    _Float16* Vt = Kh + elems;

    proj_kernel<<<3 * BATCH * (SEQ / 64), 128, 0, stream>>>(
        q_in, k_in, v_in, WQ, WK, WV, Qh, Kh, Vt);
    attn_kernel<<<BATCH * (SEQ / 64), 128, 0, stream>>>(Qh, Kh, Vt, mask, out);
}
